// PFCfC_8864812499225
// MI455X (gfx1250) — matrix operand order repaired and checked by reading, not compiled, not run on hardware
//
#include <hip/hip_runtime.h>
#include <hip/hip_bf16.h>

typedef __attribute__((ext_vector_type(16))) _Float16 v16h;
typedef __attribute__((ext_vector_type(8)))  float    v8f;

#define Bb 16
#define Tt 256
#define Ii 64
#define Hh 128
#define Kk 64
#define Uu 256
#define ZK (Ii + Hh)   // 192

union Frag16 { v16h v; unsigned u[8]; };

// ---- fast transcendentals (branch-free, single trans ops) -------------------
#define LOG2E 1.4426950408889634f

__device__ inline float fast_tanh(float x) {
#if __has_builtin(__builtin_amdgcn_tanhf)
    return __builtin_amdgcn_tanhf(x);          // v_tanh_f32
#else
    float e = __builtin_amdgcn_exp2f(x * (2.0f * LOG2E));
    return 1.0f - 2.0f * __builtin_amdgcn_rcpf(e + 1.0f);
#endif
}

__device__ inline float fast_sigmoid(float x) {
    return __builtin_amdgcn_rcpf(1.0f + __builtin_amdgcn_exp2f(-x * LOG2E));
}

// ---- fragment loaders -------------------------------------------------------
// A (16x32 f16): lanes 0-15 -> M=0..15 with K 0..7 / 16..23; lanes 16-31 same M, K 8..15 / 24..31
__device__ inline v16h load_a_lds(const _Float16* base, int row0, int stride, int kbase, int lane) {
    Frag16 a;
    int m  = lane & 15;
    int kh = lane >> 4;
    const _Float16* p = base + (row0 + m) * stride + kbase + kh * 8;
#pragma unroll
    for (int j = 0; j < 8; ++j) {
        int k0 = ((j >> 2) << 4) + ((j & 3) << 1);   // (j/4)*16 + (j%4)*2
        a.u[j] = *(const unsigned*)(p + k0);
    }
    return a.v;
}

// B (32x16 f16) from column-major weights W^T stored [n][k] (k contiguous):
// lanes 0-15 -> K 0..7 / 16..23, lanes 16-31 -> K 8..15 / 24..31; lane&15 = N
__device__ inline v16h load_b_glb(const _Float16* base, int n0, int kdim, int kbase, int lane) {
    Frag16 b;
    int n  = lane & 15;
    int kh = lane >> 4;
    const _Float16* p = base + (size_t)(n0 + n) * kdim + kbase + kh * 8;
#pragma unroll
    for (int j = 0; j < 8; ++j) {
        b.u[j] = *(const unsigned*)(p + ((j >> 2) << 4) + ((j & 3) << 1));
    }
    return b.v;
}

// ---- prep: f32 weights -> f16 transposed in workspace -----------------------
__global__ void pf_prep_kernel(const float* __restrict__ Wb,
                               const float* __restrict__ Wff1, const float* __restrict__ Wff2,
                               const float* __restrict__ Wta,  const float* __restrict__ Wtb,
                               const float* __restrict__ bff1, const float* __restrict__ bff2,
                               const float* __restrict__ bta,  const float* __restrict__ btb,
                               _Float16* __restrict__ wbT, _Float16* __restrict__ wcatT,
                               float* __restrict__ bcat) {
    int idx = blockIdx.x * blockDim.x + threadIdx.x;
    if (idx < Uu * ZK) {                       // wbT[n][k] = Wb[k][n]
        int n = idx / ZK, k = idx % ZK;
        wbT[idx] = (_Float16)Wb[k * Uu + n];
    }
    if (idx < 4 * Hh * Uu) {                   // wcatT[n][u], n = sel*128 + col
        int n = idx / Uu, u = idx % Uu;
        int sel = n >> 7, c = n & 127;
        const float* W = (sel == 0) ? Wff1 : (sel == 1) ? Wff2 : (sel == 2) ? Wta : Wtb;
        wcatT[idx] = (_Float16)W[u * Hh + c];
    }
    if (idx < 4 * Hh) {
        int sel = idx >> 7, c = idx & 127;
        const float* bs = (sel == 0) ? bff1 : (sel == 1) ? bff2 : (sel == 2) ? bta : btb;
        bcat[idx] = bs[c];
    }
}

// ---- main persistent kernel: one block per batch element --------------------
__global__ __launch_bounds__(256) void pf_cfc_kernel(
        const float* __restrict__ x,          // (B,T,I)
        const float* __restrict__ timespans,  // (B,T)
        const float* __restrict__ eps,        // (T,B,K,H)
        const _Float16* __restrict__ wbT,     // (U, ZK)
        const float* __restrict__ bb,         // (U)
        const _Float16* __restrict__ wcatT,   // (512, U)
        const float* __restrict__ bcat,       // (512)
        const float* __restrict__ noise_scale,// (H)
        float* __restrict__ out,              // (B,T,H)
        float* __restrict__ part_out,         // (B,K,H)
        float* __restrict__ logw_out) {       // (B,K)
    extern __shared__ char smem[];
    _Float16* zls  = (_Float16*)smem;                       // [K][ZK]   24576 B
    _Float16* bbk  = (_Float16*)(smem + Kk * ZK * 2);       // [K][U]    32768 B
    float*    part = (float*)(smem + Kk * ZK * 2 + Kk * Uu * 2); // [K][H] 32768 B

    const int b    = blockIdx.x;
    const int tid  = threadIdx.x;
    const int lane = tid & 31;
    const int wave = tid >> 5;

    // init: particles0 = 0 ; log_w output (constant through the scan)
    for (int e = tid; e < Kk * Hh; e += 256) {
        part[e] = 0.0f;
        int k = e >> 7, h = e & 127;
        zls[k * ZK + Ii + h] = (_Float16)0.0f;
    }
    if (tid < Kk) logw_out[b * Kk + tid] = -4.1588830833596715f;  // -log(64)
    __syncthreads();

    for (int t = 0; t < Tt; ++t) {
        const float tsv  = timespans[b * Tt + t];
        const float sqts = __builtin_amdgcn_sqrtf(tsv);

        // prefetch this step's eps slab (32 KB: K*H f32) so the phase-3 loads
        // hit cache; overlaps GEMM1+GEMM2  -> global_prefetch_b8
        {
            const char* pbase = (const char*)(eps + (size_t)(t * Bb + b) * Kk * Hh);
            __builtin_prefetch(pbase + (size_t)tid * 128, 0, 0);
        }

        // phase 1: broadcast x_t into z rows (dword-packed f16 stores)
        for (int e = tid; e < (Kk * Ii) / 2; e += 256) {
            int k = e >> 5, i2 = (e & 31) * 2;
            const float* xp = &x[(b * Tt + t) * Ii + i2];
            union { _Float16 h[2]; unsigned u; } pk;
            pk.h[0] = (_Float16)xp[0];
            pk.h[1] = (_Float16)xp[1];
            *(unsigned*)(zls + k * ZK + i2) = pk.u;
        }
        __syncthreads();

        // phase 2: GEMM1  bbk = lecun_tanh(z @ Wb + bb)   (64x192 @ 192x256)
        // 4 m-tiles x 16 n-tiles = 64 tiles -> 8 per wave (one acc live at a time)
#pragma unroll 1
        for (int i = 0; i < 8; ++i) {
            int tIdx = wave * 8 + i;
            int mt = tIdx >> 4, nt = tIdx & 15;
            v8f acc = {};
#pragma unroll 2
            for (int kc = 0; kc < 6; ++kc) {
                v16h a = load_a_lds(zls, mt * 16, ZK, kc * 32, lane);
                v16h w = load_b_glb(wbT, nt * 16, ZK, kc * 32, lane);
                acc = __builtin_amdgcn_wmma_f32_16x16x32_f16(false, a, false, w,
                                                             (short)0, acc, false, false);
            }
            int col = nt * 16 + (lane & 15);
            int kh  = lane >> 4;
            float bias = bb[col];
#pragma unroll
            for (int r = 0; r < 8; ++r) {
                int row = mt * 16 + r + 8 * kh;
                float v = 1.7159f * fast_tanh(0.666f * (acc[r] + bias));
                bbk[row * Uu + col] = (_Float16)v;
            }
        }
        __syncthreads();

        // phase 3: GEMM2 (fused ff1|ff2|ta|tb) + elementwise, tiles kept in regs
        // 4 m-tiles x 8 h-tiles = 32 groups -> 4 per wave; 4 stacked N-tiles per group
#pragma unroll 1
        for (int g = 0; g < 4; ++g) {
            int gid = wave * 4 + g;
            int mt = gid >> 3, ht = gid & 7;
            v8f a0 = {}, a1 = {}, a2 = {}, a3 = {};
#pragma unroll 2
            for (int kc = 0; kc < 8; ++kc) {
                v16h a = load_a_lds(bbk, mt * 16, Uu, kc * 32, lane);
                v16h w0 = load_b_glb(wcatT, 0 * Hh + ht * 16, Uu, kc * 32, lane);
                a0 = __builtin_amdgcn_wmma_f32_16x16x32_f16(false, a, false, w0, (short)0, a0, false, false);
                v16h w1 = load_b_glb(wcatT, 1 * Hh + ht * 16, Uu, kc * 32, lane);
                a1 = __builtin_amdgcn_wmma_f32_16x16x32_f16(false, a, false, w1, (short)0, a1, false, false);
                v16h w2 = load_b_glb(wcatT, 2 * Hh + ht * 16, Uu, kc * 32, lane);
                a2 = __builtin_amdgcn_wmma_f32_16x16x32_f16(false, a, false, w2, (short)0, a2, false, false);
                v16h w3 = load_b_glb(wcatT, 3 * Hh + ht * 16, Uu, kc * 32, lane);
                a3 = __builtin_amdgcn_wmma_f32_16x16x32_f16(false, a, false, w3, (short)0, a3, false, false);
            }
            int h  = ht * 16 + (lane & 15);
            int kh = lane >> 4;
            float bc0 = bcat[0 * Hh + h], bc1 = bcat[1 * Hh + h];
            float bc2 = bcat[2 * Hh + h], bc3 = bcat[3 * Hh + h];
            float nsc = noise_scale[h] * sqts;
#pragma unroll
            for (int r = 0; r < 8; ++r) {
                int k = mt * 16 + r + 8 * kh;          // particle index
                float ff1 = fast_tanh(a0[r] + bc0);
                float ff2 = fast_tanh(a1[r] + bc1);
                float tav = a2[r] + bc2;
                float tbv = a3[r] + bc3;
                float sig = fast_sigmoid(tav * tsv + tbv);
                float ep  = eps[((size_t)(t * Bb + b) * Kk + k) * Hh + h];
                float p   = ff1 + sig * (ff2 - ff1) + nsc * ep;
                part[k * Hh + h] = p;
                zls[k * ZK + Ii + h] = (_Float16)p;    // feeds next step's z
            }
        }
        __syncthreads();

        // phase 4: out[b,t,:] = mean over K (softmax of constant log_w == 1/K)
        if (tid < Hh) {
            float s = 0.0f;
#pragma unroll 8
            for (int k = 0; k < Kk; ++k) s += part[k * Hh + tid];
            out[((size_t)b * Tt + t) * Hh + tid] = s * (1.0f / (float)Kk);
        }
        // reduce reads `part`; next writes to `part` are after two barriers -> safe
    }

    __syncthreads();
    for (int e = tid; e < Kk * Hh; e += 256)
        part_out[(size_t)b * Kk * Hh + e] = part[e];
}

extern "C" void kernel_launch(void* const* d_in, const int* in_sizes, int n_in,
                              void* d_out, int out_size, void* d_ws, size_t ws_size,
                              hipStream_t stream) {
    const float* x    = (const float*)d_in[0];   // (B,T,I)
    const float* ts   = (const float*)d_in[1];   // (B,T)
    const float* eps  = (const float*)d_in[2];   // (T,B,K,H)
    const float* Wb   = (const float*)d_in[3];   // (192,256)
    const float* bb   = (const float*)d_in[4];   // (256)
    const float* Wff1 = (const float*)d_in[5];
    const float* bff1 = (const float*)d_in[6];
    const float* Wff2 = (const float*)d_in[7];
    const float* bff2 = (const float*)d_in[8];
    const float* Wta  = (const float*)d_in[9];
    const float* bta  = (const float*)d_in[10];
    const float* Wtb  = (const float*)d_in[11];
    const float* btb  = (const float*)d_in[12];
    const float* nsc  = (const float*)d_in[13];

    float* out      = (float*)d_out;                     // (B,T,H)
    float* part_out = out + (size_t)Bb * Tt * Hh;        // (B,K,H)
    float* logw_out = part_out + (size_t)Bb * Kk * Hh;   // (B,K)

    _Float16* wbT   = (_Float16*)d_ws;                   // 256*192 f16
    _Float16* wcatT = wbT + (size_t)Uu * ZK;             // 512*256 f16
    float*    bcat  = (float*)(wcatT + (size_t)4 * Hh * Uu); // 512 f32

    pf_prep_kernel<<<(4 * Hh * Uu + 255) / 256, 256, 0, stream>>>(
        Wb, Wff1, Wff2, Wta, Wtb, bff1, bff2, bta, btb, wbT, wcatT, bcat);

    const size_t lds_bytes = (size_t)Kk * ZK * 2 + (size_t)Kk * Uu * 2 + (size_t)Kk * Hh * 4; // 90112
    hipFuncSetAttribute((const void*)pf_cfc_kernel,
                        hipFuncAttributeMaxDynamicSharedMemorySize, (int)lds_bytes);

    pf_cfc_kernel<<<Bb, 256, lds_bytes, stream>>>(
        x, ts, eps, wbT, bb, wcatT, bcat, nsc, out, part_out, logw_out);
}
